// Net_71614284693592
// MI455X (gfx1250) — matrix operand order repaired and checked by reading, not compiled, not run on hardware
//
#include <hip/hip_runtime.h>

// ---------------------------------------------------------------------------
// MI455X / gfx1250 fused MLP (bf16 WMMA):
//   h  = x @ fc1_w^T + fc1_b ; c = relu(h*w1c+b1); c = relu(c*w2c+b2); h += c
//   h2 = relu(h @ lin2_w^T + lin2_b) ; out = h2 @ lin3_w^T + lin3_b
// GEMM: 128x256 block tile, 8 waves of 64x64 (4x4 WMMA accs),
// double-buffered LDS fed by GLOBAL_LOAD_ASYNC_TO_LDS_B128 (ASYNCcnt
// pipelined) when available, reg-staged ds_store fallback otherwise.
// ---------------------------------------------------------------------------

typedef __attribute__((ext_vector_type(16))) __bf16 v16bf;
typedef __attribute__((ext_vector_type(8)))  float  v8f;
typedef int v4i __attribute__((vector_size(16)));   // payload type of the b128 async builtin

#define BM 128
#define BN 256
#define KT 32
#define STR 40   // halfs per LDS row: 32 data + 8 pad = 80B, 16B-aligned, conflict-free

#if defined(__HIP_DEVICE_COMPILE__) &&                                   \
    __has_builtin(__builtin_amdgcn_global_load_async_to_lds_b128) &&     \
    __has_builtin(__builtin_amdgcn_s_wait_asynccnt)
#define USE_ASYNC 1
#define GLOBAL_AS __attribute__((address_space(1)))
#define LDS_AS    __attribute__((address_space(3)))
#else
#define USE_ASYNC 0
#endif

union Frag {
  uint4 q[2];
  v16bf v;
};

__device__ __forceinline__ void copy16B(const __bf16* __restrict__ g,
                                        __bf16* __restrict__ l) {
#if USE_ASYNC
  __builtin_amdgcn_global_load_async_to_lds_b128((GLOBAL_AS v4i*)g,
                                                 (LDS_AS v4i*)l, 0, 0);
#else
  *reinterpret_cast<uint4*>(l) = *reinterpret_cast<const uint4*>(g);
#endif
}

// ----------------------- fp32 -> bf16 conversion ---------------------------
__global__ __launch_bounds__(256) void cvt_f32_bf16(const float* __restrict__ s,
                                                    __bf16* __restrict__ d, int n) {
  int i = (blockIdx.x * blockDim.x + threadIdx.x) * 4;
  if (i + 4 <= n) {
    float4 f = *reinterpret_cast<const float4*>(s + i);
    union { __bf16 h[4]; uint2 u; } o;
    o.h[0] = (__bf16)f.x; o.h[1] = (__bf16)f.y;
    o.h[2] = (__bf16)f.z; o.h[3] = (__bf16)f.w;
    *reinterpret_cast<uint2*>(d + i) = o.u;
  } else {
    for (int j = i; j < n; ++j) d[j] = (__bf16)s[j];
  }
}

// ----------------------------- WMMA GEMM -----------------------------------
// C[M,N] = A[M,K] @ W[N,K]^T (+ fused epilogue); M%128==0, N%256==0, K%32==0
// MODE 0: h=gemm+bias; c=relu(h*cw1+cb1); c=relu(c*cw2+cb2); outb=h+c
// MODE 1: outb = relu(gemm+bias)
// MODE 2: outf = gemm+bias
template <int MODE>
__global__ __launch_bounds__(256) void gemm_bf16(
    const __bf16* __restrict__ A, const __bf16* __restrict__ W,
    const float* __restrict__ bias,
    const float* __restrict__ cw1, const float* __restrict__ cb1,
    const float* __restrict__ cw2, const float* __restrict__ cb2,
    __bf16* __restrict__ outb, float* __restrict__ outf,
    int M, int N, int K) {
  __shared__ alignas(16) __bf16 sA[2][BM * STR];
  __shared__ alignas(16) __bf16 sB[2][BN * STR];

  const int tid  = threadIdx.x;
  const int lane = tid & 31;
  const int wave = tid >> 5;
  const int wm   = wave >> 2;   // 0..1  (2 waves over 128 M)
  const int wn   = wave & 3;    // 0..3  (4 waves over 256 N)
  const int hl   = lane >> 4;   // half-wave group (lanes 0-15 / 16-31)
  const int l16  = lane & 15;

  const int bm = blockIdx.y * BM;
  const int bn = blockIdx.x * BN;

  // Stage one K-slice (BM x 32 of A, BN x 32 of W) into LDS buffer b.
  // 16B chunks: A = 512 (2/thread), B = 1024 (4/thread); coalesced by tid.
  auto stage = [&](int b, int k0) {
#pragma unroll
    for (int j = 0; j < 2; ++j) {
      int c = tid + 256 * j;
      int row = c >> 2, q = (c & 3) * 8;
      copy16B(A + (size_t)(bm + row) * K + k0 + q, &sA[b][row * STR + q]);
    }
#pragma unroll
    for (int j = 0; j < 4; ++j) {
      int c = tid + 256 * j;
      int row = c >> 2, q = (c & 3) * 8;
      copy16B(W + (size_t)(bn + row) * K + k0 + q, &sB[b][row * STR + q]);
    }
  };

  const v8f vzero = {0.f, 0.f, 0.f, 0.f, 0.f, 0.f, 0.f, 0.f};
  v8f acc[4][4];
#pragma unroll
  for (int i = 0; i < 4; ++i)
#pragma unroll
    for (int j = 0; j < 4; ++j) acc[i][j] = vzero;

  auto compute = [&](int b) {
    // B frags: lanes 0-15 -> K 0..15, lanes 16-31 -> K 16..31 (contiguous)
    Frag bfr[4];
#pragma unroll
    for (int nt = 0; nt < 4; ++nt) {
      const __bf16* p = &sB[b][(wn * 64 + nt * 16 + l16) * STR + hl * 16];
      bfr[nt].q[0] = *reinterpret_cast<const uint4*>(p);
      bfr[nt].q[1] = *reinterpret_cast<const uint4*>(p + 8);
    }
    // A frags: lanes 0-15 -> K 0..15, lanes 16-31 -> K 16..31 (contiguous)
#pragma unroll
    for (int mt = 0; mt < 4; ++mt) {
      Frag afr;
      const __bf16* p = &sA[b][(wm * 64 + mt * 16 + l16) * STR + hl * 16];
      afr.q[0] = *reinterpret_cast<const uint4*>(p);
      afr.q[1] = *reinterpret_cast<const uint4*>(p + 8);
#pragma unroll
      for (int nt = 0; nt < 4; ++nt) {
        acc[mt][nt] = __builtin_amdgcn_wmma_f32_16x16x32_bf16(
            false, afr.v, false, bfr[nt].v, (short)0, acc[mt][nt], false, false);
      }
    }
  };

  const int kIters = K / KT;
#if USE_ASYNC
  stage(0, 0);
  for (int kt = 0; kt < kIters; ++kt) {
    if (kt + 1 < kIters) {
      stage((kt + 1) & 1, (kt + 1) * KT);           // 6 async loads / wave
      __builtin_amdgcn_s_wait_asynccnt(6);          // tile kt resident
    } else {
      __builtin_amdgcn_s_wait_asynccnt(0);
    }
    __syncthreads();
    compute(kt & 1);
    __syncthreads();
  }
#else
  for (int kt = 0; kt < kIters; ++kt) {
    stage(kt & 1, kt * KT);
    __syncthreads();
    compute(kt & 1);
    __syncthreads();
  }
#endif

  // Epilogue. C/D layout: VGPR r -> row (r + hl*8), lane%16 -> col.
#pragma unroll
  for (int mt = 0; mt < 4; ++mt) {
    const int row0 = bm + wm * 64 + mt * 16 + hl * 8;
#pragma unroll
    for (int nt = 0; nt < 4; ++nt) {
      const int col = bn + wn * 64 + nt * 16 + l16;
      const float bs = bias[col];
      float m1 = 0.f, a1 = 0.f, m2 = 0.f, a2 = 0.f;
      if (MODE == 0) {
        m1 = cw1[col * 3 + 1]; a1 = cb1[col];   // center tap of 3-tap conv
        m2 = cw2[col * 3 + 1]; a2 = cb2[col];
      }
#pragma unroll
      for (int r = 0; r < 8; ++r) {
        float v = acc[mt][nt][r] + bs;
        if (MODE == 0) {
          float c = fmaxf(v * m1 + a1, 0.f);
          c = fmaxf(c * m2 + a2, 0.f);
          outb[(size_t)(row0 + r) * N + col] = (__bf16)(v + c);
        } else if (MODE == 1) {
          outb[(size_t)(row0 + r) * N + col] = (__bf16)fmaxf(v, 0.f);
        } else {
          outf[(size_t)(row0 + r) * N + col] = v;
        }
      }
    }
  }
}

// ------------------------------- launcher ----------------------------------
extern "C" void kernel_launch(void* const* d_in, const int* in_sizes, int n_in,
                              void* d_out, int out_size, void* d_ws, size_t ws_size,
                              hipStream_t stream) {
  const int B = 2048, IN = 4096, HID = 4096, OUT = 1024;

  const float* x      = (const float*)d_in[0];
  const float* fc1_w  = (const float*)d_in[1];
  const float* fc1_b  = (const float*)d_in[2];
  const float* w1     = (const float*)d_in[3];
  const float* b1     = (const float*)d_in[4];
  const float* w2     = (const float*)d_in[5];
  const float* b2     = (const float*)d_in[6];
  const float* lin2_w = (const float*)d_in[7];
  const float* lin2_b = (const float*)d_in[8];
  const float* lin3_w = (const float*)d_in[9];
  const float* lin3_b = (const float*)d_in[10];

  char* ws = (char*)d_ws;
  size_t off = 0;
  auto alloc = [&](size_t bytes) -> char* {
    char* p = ws + off;
    off = (off + bytes + 255) & ~(size_t)255;
    return p;
  };
  __bf16* xb     = (__bf16*)alloc((size_t)B * IN * 2);
  __bf16* fc1wb  = (__bf16*)alloc((size_t)HID * IN * 2);
  __bf16* lin2wb = (__bf16*)alloc((size_t)HID * HID * 2);
  __bf16* lin3wb = (__bf16*)alloc((size_t)OUT * HID * 2);
  __bf16* h1     = (__bf16*)alloc((size_t)B * HID * 2);
  __bf16* h2     = (__bf16*)alloc((size_t)B * HID * 2);

  auto cvt = [&](const float* s, __bf16* d, int n) {
    int blocks = (n / 4 + 255) / 256;
    hipLaunchKernelGGL(cvt_f32_bf16, dim3(blocks), dim3(256), 0, stream, s, d, n);
  };
  cvt(x,      xb,     B * IN);
  cvt(fc1_w,  fc1wb,  HID * IN);
  cvt(lin2_w, lin2wb, HID * HID);
  cvt(lin3_w, lin3wb, OUT * HID);

  dim3 blk(256);
  // fc1 + fused scalar-ResConv epilogue -> h1 (bf16)
  hipLaunchKernelGGL((gemm_bf16<0>), dim3(HID / BN, B / BM), blk, 0, stream,
                     xb, fc1wb, fc1_b, w1, b1, w2, b2,
                     h1, (float*)nullptr, B, HID, IN);
  // lin2 + bias + relu -> h2 (bf16)
  hipLaunchKernelGGL((gemm_bf16<1>), dim3(HID / BN, B / BM), blk, 0, stream,
                     h1, lin2wb, lin2_b,
                     (const float*)nullptr, (const float*)nullptr,
                     (const float*)nullptr, (const float*)nullptr,
                     h2, (float*)nullptr, B, HID, HID);
  // lin3 + bias -> d_out (fp32)
  hipLaunchKernelGGL((gemm_bf16<2>), dim3(OUT / BN, B / BM), blk, 0, stream,
                     h2, lin3wb, lin3_b,
                     (const float*)nullptr, (const float*)nullptr,
                     (const float*)nullptr, (const float*)nullptr,
                     (__bf16*)nullptr, (float*)d_out, B, OUT, HID);
}
